// TtBloomAttention_39367670235204
// MI455X (gfx1250) — compile-verified, matrix operand order checked by reading, not run on hardware
//
#include <hip/hip_runtime.h>

#define S_LEN 2048
#define HDIM  4096
#define NHEAD 32
#define HEADD 128

typedef __attribute__((ext_vector_type(16))) __bf16        v16bf;
typedef __attribute__((ext_vector_type(8)))  float         v8f;
typedef __attribute__((ext_vector_type(4)))  unsigned int  u32x4;
typedef __attribute__((ext_vector_type(2)))  unsigned int  u32x2;
typedef __attribute__((ext_vector_type(4)))  float         f32x4;

// fp32 -> bf16, round-half-up (bias ~half-ulp of bf16; negligible vs bf16 quant)
static __device__ __forceinline__ unsigned short f2bf(float x) {
  return (unsigned short)((__float_as_uint(x) + 0x8000u) >> 16);
}

// Pack two floats to packed bf16 in 3 VALU: 2x v_add + 1x v_perm_b32.
// perm byte pool: bytes 0-3 = src1(a), bytes 4-7 = src0(b);
// result = { a[2],a[3], b[2],b[3] } = lo16 from a.hi16, hi16 from b.hi16.
static __device__ __forceinline__ unsigned int pkbf(float x, float y) {
  unsigned int a = __float_as_uint(x) + 0x8000u;
  unsigned int b = __float_as_uint(y) + 0x8000u;
  return __builtin_amdgcn_perm(b, a, 0x07060302u);
}

static __device__ __forceinline__ v8f vzero() {
  v8f z = {0.f, 0.f, 0.f, 0.f, 0.f, 0.f, 0.f, 0.f};
  return z;
}

// Load one 16x32 bf16 WMMA operand fragment from LDS.
// rowbase = LDS byte address of this lane's row (row m = lane&15 for A,
// column n = lane&15 for B) at the start of the 32-wide k-chunk.
// Lane layout per ISA: elems 0..7 -> k = hi*8+0..7, elems 8..15 -> k = 16+hi*8+0..7
static __device__ __forceinline__ v16bf ld_frag(const char* rowbase, int hi) {
  union { v16bf v; u32x4 q[2]; } f;
  f.q[0] = *(const u32x4*)(rowbase + hi * 16);
  f.q[1] = *(const u32x4*)(rowbase + 32 + hi * 16);
  return f.v;
}

// reduce across the 16 lanes of each wave-half (wave32)
static __device__ __forceinline__ float half_max(float x) {
  x = fmaxf(x, __shfl_xor(x, 1, 32));
  x = fmaxf(x, __shfl_xor(x, 2, 32));
  x = fmaxf(x, __shfl_xor(x, 4, 32));
  x = fmaxf(x, __shfl_xor(x, 8, 32));
  return x;
}
static __device__ __forceinline__ float half_sum(float x) {
  x += __shfl_xor(x, 1, 32);
  x += __shfl_xor(x, 2, 32);
  x += __shfl_xor(x, 4, 32);
  x += __shfl_xor(x, 8, 32);
  return x;
}

// ---------------------------------------------------------------------------
// Kernel 1: fused QKV projection.  X(4096x4096 f32) @ Wqkv^T(4096x12288) + b
// -> scatter bf16 into per-head Q/K/V [b][nh][s][hd].
// 128x128 block tile, 8 waves (32x64 sub-tile each), double-buffered LDS with
// register staging: next tile's global loads issue before current compute.
// ---------------------------------------------------------------------------
__global__ __launch_bounds__(256) void gemm_qkv(
    const float* __restrict__ X, const float* __restrict__ W,
    const float* __restrict__ bias,
    unsigned short* __restrict__ Qo, unsigned short* __restrict__ Ko,
    unsigned short* __restrict__ Vo)
{
  __shared__ __align__(16) unsigned short As[2][128 * 40]; // 32 bf16/row + pad
  __shared__ __align__(16) unsigned short Bs[2][128 * 40];
  const int tid  = threadIdx.x;
  const int lane = tid & 31, wave = tid >> 5;
  const int l15  = lane & 15, hi = lane >> 4;
  const int wrow = wave & 3,  wcol = wave >> 2;
  const int mbase = blockIdx.x * 128;
  const int nbase = blockIdx.y * 128;

  // per-thread staging coordinates: 4 chunks of float4 per matrix
  int srow[4], skc[4];
  for (int c = 0; c < 4; ++c) {
    int ch = tid + 256 * c;
    srow[c] = ch >> 3;
    skc[c]  = ch & 7;
  }

  v8f acc[2][4];
  for (int a = 0; a < 2; ++a)
    for (int c = 0; c < 4; ++c) acc[a][c] = vzero();

  f32x4 ra[4], rb[4];
  // preload tile 0
  for (int c = 0; c < 4; ++c) {
    ra[c] = *(const f32x4*)(X + (size_t)(mbase + srow[c]) * HDIM + skc[c] * 4);
    rb[c] = *(const f32x4*)(W + (size_t)(nbase + srow[c]) * HDIM + skc[c] * 4);
  }
  for (int c = 0; c < 4; ++c) {
    u32x2 pa, pb;
    pa.x = pkbf(ra[c].x, ra[c].y); pa.y = pkbf(ra[c].z, ra[c].w);
    pb.x = pkbf(rb[c].x, rb[c].y); pb.y = pkbf(rb[c].z, rb[c].w);
    *(u32x2*)((char*)As[0] + srow[c] * 80 + skc[c] * 8) = pa;
    *(u32x2*)((char*)Bs[0] + srow[c] * 80 + skc[c] * 8) = pb;
  }

  int cur = 0;
  for (int kb = 0; kb < HDIM; kb += 32) {
    const bool more = (kb + 32) < HDIM;
    if (more) {  // issue next tile's global loads before computing
      for (int c = 0; c < 4; ++c) {
        ra[c] = *(const f32x4*)(X + (size_t)(mbase + srow[c]) * HDIM + kb + 32 + skc[c] * 4);
        rb[c] = *(const f32x4*)(W + (size_t)(nbase + srow[c]) * HDIM + kb + 32 + skc[c] * 4);
      }
    }
    __syncthreads();  // publish LDS[cur]; fence LDS[cur^1] before rewrite
    v16bf af[2], bfr[4];
    for (int a = 0; a < 2; ++a)
      af[a] = ld_frag((const char*)As[cur] + (wrow * 32 + a * 16 + l15) * 80, hi);
    for (int c = 0; c < 4; ++c)
      bfr[c] = ld_frag((const char*)Bs[cur] + (wcol * 64 + c * 16 + l15) * 80, hi);
    for (int c = 0; c < 4; ++c)
      for (int a = 0; a < 2; ++a)
        acc[a][c] = __builtin_amdgcn_wmma_f32_16x16x32_bf16(
            false, af[a], false, bfr[c], (short)0, acc[a][c], false, false);
    if (more) {
      const int nxt = cur ^ 1;
      for (int c = 0; c < 4; ++c) {
        u32x2 pa, pb;
        pa.x = pkbf(ra[c].x, ra[c].y); pa.y = pkbf(ra[c].z, ra[c].w);
        pb.x = pkbf(rb[c].x, rb[c].y); pb.y = pkbf(rb[c].z, rb[c].w);
        *(u32x2*)((char*)As[nxt] + srow[c] * 80 + skc[c] * 8) = pa;
        *(u32x2*)((char*)Bs[nxt] + srow[c] * 80 + skc[c] * 8) = pb;
      }
    }
    cur ^= 1;
  }

  // epilogue: +bias, scatter into Q/K/V as bf16
  for (int c = 0; c < 4; ++c) {
    int o = nbase + wcol * 64 + c * 16 + l15;        // column in 3H
    float bia = bias[o];
    int nh = o / 384, trip = (o / 128) % 3, hd = o % 128;
    unsigned short* dst = (trip == 0) ? Qo : ((trip == 1) ? Ko : Vo);
    for (int a = 0; a < 2; ++a) {
      for (int r = 0; r < 8; ++r) {
        int row = mbase + wrow * 32 + a * 16 + r + 8 * hi;  // token index
        int b = row >> 11, s = row & 2047;
        float val = acc[a][c][r] + bia;
        dst[(((size_t)(b * NHEAD + nh)) * S_LEN + s) * HEADD + hd] = f2bf(val);
      }
    }
  }
}

// ---------------------------------------------------------------------------
// Kernel 2: fused flash attention per (b, head, 128-row Q block).
// S = QK^T * inv_norm + alibi, causal mask, online softmax, O += P V.
// 8 waves x 16 score rows each. K-tile LDS reused for bf16 P tile.
// ---------------------------------------------------------------------------
__global__ __launch_bounds__(256) void attn(
    const unsigned short* __restrict__ Qg, const unsigned short* __restrict__ Kg,
    const unsigned short* __restrict__ Vg, const float* __restrict__ alibi,
    unsigned short* __restrict__ ctx)
{
  __shared__ __align__(16) unsigned short Qs [128 * 136]; // [qrow][hd],  272B rows
  __shared__ __align__(16) unsigned short KPs[128 * 136]; // K tile, then P tile
  __shared__ __align__(16) unsigned short Vts[128 * 136]; // V transposed [hd][krow]
  __shared__ __align__(16) float Als[S_LEN];

  const int tid  = threadIdx.x;
  const int lane = tid & 31, wave = tid >> 5;
  const int l15  = lane & 15, hi = lane >> 4;
  const int qb = blockIdx.x, nh = blockIdx.y, b = blockIdx.z;
  const int qbase = qb * 128;
  const size_t headoff = ((size_t)(b * NHEAD + nh)) * S_LEN * HEADD;
  const unsigned short* Qh = Qg + headoff;
  const unsigned short* Kh = Kg + headoff;
  const unsigned short* Vh = Vg + headoff;
  const float* alh = alibi + (size_t)(b * NHEAD + nh) * S_LEN;

  // alibi row -> LDS
  for (int c = 0; c < 2; ++c) {
    int ch = tid + 256 * c;
    *(f32x4*)(Als + ch * 4) = *(const f32x4*)(alh + ch * 4);
  }
  // Q tile -> LDS
  for (int c = 0; c < 4; ++c) {
    int ch = tid + 256 * c;
    int row = ch >> 3, k8 = ch & 7;
    *(u32x4*)((char*)Qs + row * 272 + k8 * 16) =
        *(const u32x4*)(Qh + (size_t)(qbase + row) * HEADD + k8 * 8);
  }
  __syncthreads();

  v16bf qf[4];
  for (int kk = 0; kk < 4; ++kk)
    qf[kk] = ld_frag((const char*)Qs + (wave * 16 + l15) * 272 + kk * 64, hi);

  float mrow[8], lrow[8];
  v8f oacc[8];
  for (int r = 0; r < 8; ++r) { mrow[r] = -3.0e38f; lrow[r] = 0.f; }
  for (int j = 0; j < 8; ++j) oacc[j] = vzero();

  const float INVN = 0.08838834764831845f; // 1/sqrt(128)
  const int nkb = qb + 1;                  // causal: only k-blocks <= q-block
  for (int kb = 0; kb < nkb; ++kb) {
    const int kbase = kb * 128;
    __syncthreads(); // previous PV fragment reads of KPs/Vts complete
    // stage K (row-major) and V (transposed) tiles
    for (int c = 0; c < 4; ++c) {
      int ch = tid + 256 * c;
      int row = ch >> 3, k8 = ch & 7;
      *(u32x4*)((char*)KPs + row * 272 + k8 * 16) =
          *(const u32x4*)(Kh + (size_t)(kbase + row) * HEADD + k8 * 8);
      u32x4 vv = *(const u32x4*)(Vh + (size_t)(kbase + row) * HEADD + k8 * 8);
      for (int e = 0; e < 8; ++e) {
        unsigned short ev = (unsigned short)((vv[e >> 1] >> ((e & 1) * 16)) & 0xFFFFu);
        Vts[(k8 * 8 + e) * 136 + row] = ev;
      }
    }
    __syncthreads();

    // S = Q K^T : B-fragment column n = K row (hd contiguous in KPs rows)
    v8f sacc[8];
    for (int j = 0; j < 8; ++j) {
      sacc[j] = vzero();
      for (int kk = 0; kk < 4; ++kk) {
        v16bf kf = ld_frag((const char*)KPs + (j * 16 + l15) * 272 + kk * 64, hi);
        sacc[j] = __builtin_amdgcn_wmma_f32_16x16x32_bf16(
            false, qf[kk], false, kf, (short)0, sacc[j], false, false);
      }
    }

    // scale + alibi + causal mask; per-lane row-tile max
    float tmax[8];
    for (int r = 0; r < 8; ++r) tmax[r] = -3.0e38f;
    const bool diag = (kb == qb);
    for (int j = 0; j < 8; ++j) {
      int kc = kbase + j * 16 + l15;
      float al = Als[kc];
      for (int r = 0; r < 8; ++r) {
        float v = sacc[j][r] * INVN + al;
        if (diag) {
          int qr = qbase + wave * 16 + r + 8 * hi;
          if (kc > qr) v = -100000.0f;
        }
        sacc[j][r] = v;
        tmax[r] = fmaxf(tmax[r], v);
      }
    }
    float alpha[8];
    for (int r = 0; r < 8; ++r) {
      float tm   = half_max(tmax[r]);
      float mnew = fmaxf(mrow[r], tm);
      alpha[r] = __expf(mrow[r] - mnew);
      mrow[r]  = mnew;
    }
    __syncthreads(); // all waves done reading K tile out of KPs
    float tsum[8];
    for (int r = 0; r < 8; ++r) tsum[r] = 0.f;
    for (int j = 0; j < 8; ++j) {
      for (int r = 0; r < 8; ++r) {
        float p = __expf(sacc[j][r] - mrow[r]);
        tsum[r] += p;
        KPs[(wave * 16 + r + 8 * hi) * 136 + j * 16 + l15] = f2bf(p);
      }
    }
    for (int r = 0; r < 8; ++r)
      lrow[r] = lrow[r] * alpha[r] + half_sum(tsum[r]);
    for (int j = 0; j < 8; ++j)
      for (int r = 0; r < 8; ++r)
        oacc[j][r] *= alpha[r];
    __syncthreads(); // P tile fully written

    // O += P V : A = P rows, B columns = hd (k contiguous along Vts rows)
    v16bf pf[4];
    for (int kk = 0; kk < 4; ++kk)
      pf[kk] = ld_frag((const char*)KPs + (wave * 16 + l15) * 272 + kk * 64, hi);
    for (int j = 0; j < 8; ++j)
      for (int kk = 0; kk < 4; ++kk) {
        v16bf vf = ld_frag((const char*)Vts + (j * 16 + l15) * 272 + kk * 64, hi);
        oacc[j] = __builtin_amdgcn_wmma_f32_16x16x32_bf16(
            false, pf[kk], false, vf, (short)0, oacc[j], false, false);
      }
  }

  // normalize and write ctx (token-major [b][s][nh*HD+hd], bf16)
  for (int j = 0; j < 8; ++j) {
    int hd = j * 16 + l15;
    for (int r = 0; r < 8; ++r) {
      int row = qbase + wave * 16 + r + 8 * hi;
      float o = oacc[j][r] / lrow[r];
      ctx[((size_t)(b * S_LEN + row)) * HDIM + nh * HEADD + hd] = f2bf(o);
    }
  }
}

// ---------------------------------------------------------------------------
// Kernel 3: dense projection + bias + residual, fp32 out.
// ctx(bf16 4096x4096) @ Wd^T(4096x4096 f32->bf16), double-buffered like k1.
// ---------------------------------------------------------------------------
__global__ __launch_bounds__(256) void gemm_dense(
    const unsigned short* __restrict__ A, const float* __restrict__ W,
    const float* __restrict__ bias, const float* __restrict__ resid,
    float* __restrict__ out)
{
  __shared__ __align__(16) unsigned short As[2][128 * 40];
  __shared__ __align__(16) unsigned short Bs[2][128 * 40];
  const int tid  = threadIdx.x;
  const int lane = tid & 31, wave = tid >> 5;
  const int l15  = lane & 15, hi = lane >> 4;
  const int wrow = wave & 3,  wcol = wave >> 2;
  const int mbase = blockIdx.x * 128;
  const int nbase = blockIdx.y * 128;

  // A staging: 2 chunks of 8 bf16; B staging: 4 chunks of float4
  int arow[2], akc[2], brow[4], bkc[4];
  for (int c = 0; c < 2; ++c) {
    int ch = tid + 256 * c;
    arow[c] = ch >> 2; akc[c] = ch & 3;
  }
  for (int c = 0; c < 4; ++c) {
    int ch = tid + 256 * c;
    brow[c] = ch >> 3; bkc[c] = ch & 7;
  }

  v8f acc[2][4];
  for (int a = 0; a < 2; ++a)
    for (int c = 0; c < 4; ++c) acc[a][c] = vzero();

  u32x4 rA[2]; f32x4 rB[4];
  for (int c = 0; c < 2; ++c)
    rA[c] = *(const u32x4*)(A + (size_t)(mbase + arow[c]) * HDIM + akc[c] * 8);
  for (int c = 0; c < 4; ++c)
    rB[c] = *(const f32x4*)(W + (size_t)(nbase + brow[c]) * HDIM + bkc[c] * 4);
  for (int c = 0; c < 2; ++c)
    *(u32x4*)((char*)As[0] + arow[c] * 80 + akc[c] * 16) = rA[c];
  for (int c = 0; c < 4; ++c) {
    u32x2 pb;
    pb.x = pkbf(rB[c].x, rB[c].y); pb.y = pkbf(rB[c].z, rB[c].w);
    *(u32x2*)((char*)Bs[0] + brow[c] * 80 + bkc[c] * 8) = pb;
  }

  int cur = 0;
  for (int kb = 0; kb < HDIM; kb += 32) {
    const bool more = (kb + 32) < HDIM;
    if (more) {
      for (int c = 0; c < 2; ++c)
        rA[c] = *(const u32x4*)(A + (size_t)(mbase + arow[c]) * HDIM + kb + 32 + akc[c] * 8);
      for (int c = 0; c < 4; ++c)
        rB[c] = *(const f32x4*)(W + (size_t)(nbase + brow[c]) * HDIM + kb + 32 + bkc[c] * 4);
    }
    __syncthreads();
    v16bf af[2], bfr[4];
    for (int a = 0; a < 2; ++a)
      af[a] = ld_frag((const char*)As[cur] + (wrow * 32 + a * 16 + l15) * 80, hi);
    for (int c = 0; c < 4; ++c)
      bfr[c] = ld_frag((const char*)Bs[cur] + (wcol * 64 + c * 16 + l15) * 80, hi);
    for (int c = 0; c < 4; ++c)
      for (int a = 0; a < 2; ++a)
        acc[a][c] = __builtin_amdgcn_wmma_f32_16x16x32_bf16(
            false, af[a], false, bfr[c], (short)0, acc[a][c], false, false);
    if (more) {
      const int nxt = cur ^ 1;
      for (int c = 0; c < 2; ++c)
        *(u32x4*)((char*)As[nxt] + arow[c] * 80 + akc[c] * 16) = rA[c];
      for (int c = 0; c < 4; ++c) {
        u32x2 pb;
        pb.x = pkbf(rB[c].x, rB[c].y); pb.y = pkbf(rB[c].z, rB[c].w);
        *(u32x2*)((char*)Bs[nxt] + brow[c] * 80 + bkc[c] * 8) = pb;
      }
    }
    cur ^= 1;
  }

  for (int c = 0; c < 4; ++c) {
    int o = nbase + wcol * 64 + c * 16 + l15;
    float bia = bias[o];
    for (int a = 0; a < 2; ++a) {
      for (int r = 0; r < 8; ++r) {
        int row = mbase + wrow * 32 + a * 16 + r + 8 * hi;
        size_t idx = (size_t)row * HDIM + o;
        out[idx] = acc[a][c][r] + bia + resid[idx];
      }
    }
  }
}

// ---------------------------------------------------------------------------
extern "C" void kernel_launch(void* const* d_in, const int* in_sizes, int n_in,
                              void* d_out, int out_size, void* d_ws, size_t ws_size,
                              hipStream_t stream) {
  (void)in_sizes; (void)n_in; (void)out_size; (void)ws_size;
  const float* hs    = (const float*)d_in[0]; // hidden_states (B,S,H)
  const float* res   = (const float*)d_in[1]; // residual
  const float* alibi = (const float*)d_in[2]; // (B*NH,1,S)
  // d_in[3] attention_mask: causal triu — derived analytically, unused
  const float* Wqkv  = (const float*)d_in[4]; // (3H,H)
  const float* bqkv  = (const float*)d_in[5];
  const float* Wd    = (const float*)d_in[6]; // (H,H)
  const float* bd    = (const float*)d_in[7];
  float* out = (float*)d_out;

  const size_t QKV_ELEMS = (size_t)2 * NHEAD * S_LEN * HEADD; // 16,777,216
  unsigned short* Qw = (unsigned short*)d_ws;
  unsigned short* Kw = Qw + QKV_ELEMS;
  unsigned short* Vw = Kw + QKV_ELEMS;
  unsigned short* Cw = Vw + QKV_ELEMS; // ctx bf16 (B*S, H)

  gemm_qkv  <<<dim3(32, 96),        256, 0, stream>>>(hs, Wqkv, bqkv, Qw, Kw, Vw);
  attn      <<<dim3(16, NHEAD, 2),  256, 0, stream>>>(Qw, Kw, Vw, alibi, Cw);
  gemm_dense<<<dim3(32, 32),        256, 0, stream>>>(Cw, Wd, bd, res, out);
}
